// MultiheadSelfAttention_86552180949808
// MI455X (gfx1250) — matrix operand order repaired and checked by reading, not compiled, not run on hardware
//
#include <hip/hip_runtime.h>

// MI455X / gfx1250 multi-head self-attention.
// bf16 WMMA for all matmuls, double-buffered TDM (tensor_load_to_lds) staging
// of K/V tiles overlapped with compute, DPP16 butterfly softmax reductions.
// B=1, S=4096, D=1024, H=16, HD=64. Requires ~48 MiB workspace.

#define S_  4096
#define D_  1024
#define H_  16
#define HD_ 64

typedef unsigned short u16;
typedef __attribute__((ext_vector_type(16))) __bf16          v16bf;
typedef __attribute__((ext_vector_type(16))) unsigned short  v16u;
typedef __attribute__((ext_vector_type(8)))  float           v8f;
typedef __attribute__((ext_vector_type(4)))  unsigned int    v4u;
typedef __attribute__((ext_vector_type(8)))  int             v8i;
typedef __attribute__((ext_vector_type(4)))  int             v4i;

// ---- helpers -------------------------------------------------------------

__device__ __forceinline__ u16 f2bf(float f) {            // f32 -> bf16 RNE
  unsigned u = __builtin_bit_cast(unsigned, f);
  u += 0x7FFFu + ((u >> 16) & 1u);
  return (u16)(u >> 16);
}

// A-matrix fragment, 16x32 bf16 (M x K). Lane L: row m = L&15, half h = L>>4.
__device__ __forceinline__ v16bf load_frag_a(const u16* p, int lda, int lane) {
  const int m = lane & 15, h = lane >> 4;
  const u16* r = p + m * lda + h * 8;
  v16u v;
  reinterpret_cast<uint4*>(&v)[0] = *reinterpret_cast<const uint4*>(r);
  reinterpret_cast<uint4*>(&v)[1] = *reinterpret_cast<const uint4*>(r + 16);
  return __builtin_bit_cast(v16bf, v);
}

// B-matrix fragment, 32x16 bf16 (K x N). Source row-major [N][K].
__device__ __forceinline__ v16bf load_frag_b(const u16* p, int ldb, int lane) {
  const int n = lane & 15, h = lane >> 4;
  const u16* r = p + n * ldb + h * 8;
  v16u v;
  reinterpret_cast<uint4*>(&v)[0] = reinterpret_cast<const uint4*>(r)[0];
  reinterpret_cast<uint4*>(&v)[1] = *reinterpret_cast<const uint4*>(r + 16);
  return __builtin_bit_cast(v16bf, v);
}

__device__ __forceinline__ v8f wmma_bf16(v16bf a, v16bf b, v8f c) {
  return __builtin_amdgcn_wmma_f32_16x16x32_bf16(false, a, false, b,
                                                 (short)0, c, false, false);
}

// ---- DPP16 butterfly reduction across the 16-lane half that holds a row ---
template <int CTRL>
__device__ __forceinline__ float dppf(float x) {
  return __builtin_bit_cast(
      float, __builtin_amdgcn_mov_dpp(__builtin_bit_cast(int, x), CTRL, 0xF,
                                      0xF, true));
}
__device__ __forceinline__ float red16_max(float x) {
  x = fmaxf(x, dppf<0xB1>(x));    // quad_perm(1,0,3,2)  : xor 1
  x = fmaxf(x, dppf<0x4E>(x));    // quad_perm(2,3,0,1)  : xor 2
  x = fmaxf(x, dppf<0x141>(x));   // ROW_HALF_MIRROR     : xor 7
  x = fmaxf(x, dppf<0x140>(x));   // ROW_MIRROR          : xor 15
  return x;
}
__device__ __forceinline__ float red16_sum(float x) {
  x += dppf<0xB1>(x);
  x += dppf<0x4E>(x);
  x += dppf<0x141>(x);
  x += dppf<0x140>(x);
  return x;
}

// ---- TDM: 2D tile (tile_x contiguous elems x tile_y rows) global -> LDS ---
// bf16 elements. Optional padding: +4 DWORDs every 32 DWORDs (row 64 elems ->
// LDS stride 72 elems = 144 B, bank-conflict-free for fragment reads).
__device__ __forceinline__ void tdm_load_2d(unsigned lds_byte, const void* g,
                                            unsigned tile_x, unsigned tile_y,
                                            unsigned stride0, unsigned dim0,
                                            unsigned dim1, bool pad) {
  unsigned long long ga = (unsigned long long)(uintptr_t)g;
  v4u g0;
  g0[0] = 1u;                                   // count=1, user descriptor
  g0[1] = lds_byte;                             // LDS byte address
  g0[2] = (unsigned)ga;                         // global_addr[31:0]
  g0[3] = (unsigned)((ga >> 32) & 0x01FFFFFFu) | (2u << 30);   // type=2
  unsigned d0 = (1u << 16);                     // data_size = 2 bytes
  if (pad) d0 |= (1u << 20) | (4u << 22) | (3u << 25);  // pad 4 dw / 32 dw
  v8i g1;
  g1[0] = (int)d0;                              // wg_mask=0
  g1[1] = (int)((dim0 & 0xFFFFu) << 16);        // tensor_dim0[15:0]
  g1[2] = (int)(((dim0 >> 16) & 0xFFFFu) | ((dim1 & 0xFFFFu) << 16));
  g1[3] = (int)(((dim1 >> 16) & 0xFFFFu) | ((tile_x & 0xFFFFu) << 16));
  g1[4] = (int)(tile_y & 0xFFFFu);              // tile_dim1 (tile_dim2=0)
  g1[5] = (int)stride0;                         // tensor_dim0_stride[31:0]
  g1[6] = 0;
  g1[7] = 0;
  v4i z4 = {};
#if defined(__clang_major__) && __clang_major__ >= 23
  v8i z8 = {};
  __builtin_amdgcn_tensor_load_to_lds(g0, g1, z4, z4, z8, 0);
#else
  __builtin_amdgcn_tensor_load_to_lds(g0, g1, z4, z4, 0);
#endif
}

// ---- stage 0: f32 -> bf16 conversion ------------------------------------

__global__ void cvt_f32_bf16(const float* __restrict__ in,
                             u16* __restrict__ out, int n) {
  int i = blockIdx.x * blockDim.x + threadIdx.x;
  int stride = gridDim.x * blockDim.x;
  for (; i < n; i += stride) out[i] = f2bf(in[i]);
}

// ---- GEMM  C[4096,1024] = A[4096,1024] * Wrows[1024,1024]^T --------------
// MODE 0: bf16 row-major (Q,K) | MODE 1: bf16 transposed (Vt) | MODE 2: f32.

template <int MODE>
__global__ void gemm16(const u16* __restrict__ A,
                       const u16* __restrict__ Wr,
                       void* __restrict__ Cout) {
  const int lane = threadIdx.x & 31;
  const int wave = threadIdx.x >> 5;
  const int job = blockIdx.x * 8 + wave;      // 4096 jobs
  const int mtile = job >> 4;                 // 0..255
  const int nstrip = job & 15;                // 0..15
  const u16* Abase = A + (size_t)mtile * 16 * D_;

  v8f acc[4] = {{}, {}, {}, {}};
  for (int k0 = 0; k0 < D_; k0 += 32) {
    v16bf a = load_frag_a(Abase + k0, D_, lane);
#pragma unroll
    for (int j = 0; j < 4; ++j) {
      v16bf b = load_frag_b(Wr + (size_t)(nstrip * 64 + j * 16) * D_ + k0, D_, lane);
      acc[j] = wmma_bf16(a, b, acc[j]);
    }
  }

  const int half = lane >> 4, n0 = lane & 15;
#pragma unroll
  for (int j = 0; j < 4; ++j) {
#pragma unroll
    for (int v = 0; v < 8; ++v) {
      const int row = mtile * 16 + v + 8 * half;
      const int col = nstrip * 64 + j * 16 + n0;
      if (MODE == 0) {
        ((u16*)Cout)[(size_t)row * D_ + col] = f2bf(acc[j][v]);
      } else if (MODE == 1) {
        ((u16*)Cout)[(size_t)col * S_ + row] = f2bf(acc[j][v]);
      } else {
        ((float*)Cout)[(size_t)row * D_ + col] = acc[j][v];
      }
    }
  }
}

// ---- causal flash attention ----------------------------------------------
// Block: 8 waves / 128 query rows of one head. Key loop in 64-wide chunks.
// TDM double-buffers K (64x64) and Vt (64x64) tiles in padded LDS: chunk c+1
// is DMA'd while chunk c is computed. Waves past their causal range idle but
// participate in barriers. launch_bounds(256,1) to avoid VGPR spills.

#define LDP 72   // padded LDS row stride in elems (64 + 8)

__global__ void __launch_bounds__(256, 1)
flash_attn(const u16* __restrict__ Q,
           const u16* __restrict__ K,
           const u16* __restrict__ Vt,
           u16* __restrict__ Ab) {
  __shared__ __align__(16) u16 Ks[2][64 * LDP];    // [t][hd]   (padded)
  __shared__ __align__(16) u16 Vs[2][64 * LDP];    // [hd][t]   (padded)
  __shared__ __align__(16) u16 Pt[8][16 * LDP];    // per-wave P tile 16x64

  const int lane = threadIdx.x & 31;
  const int wave = threadIdx.x >> 5;
  const int h = blockIdx.x & 15;
  const int qb = blockIdx.x >> 4;
  const int qr0 = qb * 128 + wave * 16;
  const int half = lane >> 4, n0 = lane & 15;
  const float NEG = -1.0e30f;
  const float SCL = 0.125f * 1.44269504f;          // 1/sqrt(64) * log2(e)

  // Q fragments (contraction dim = hd, split 0..31 / 32..63)
  const u16* qbase = Q + (size_t)qr0 * D_ + h * HD_;
  v16bf aq0 = load_frag_a(qbase, D_, lane);
  v16bf aq1 = load_frag_a(qbase + 32, D_, lane);

  float mrow[8], lrow[8];
#pragma unroll
  for (int v = 0; v < 8; ++v) { mrow[v] = NEG; lrow[v] = 0.0f; }
  v8f o[4] = {{}, {}, {}, {}};

  const u16* Kh = K + h * HD_;                     // K tile column base
  const u16* Vh = Vt + (size_t)(h * HD_) * S_;     // Vt tile row base
  unsigned ldsK[2] = {(unsigned)(uintptr_t)&Ks[0][0],
                      (unsigned)(uintptr_t)&Ks[1][0]};
  unsigned ldsV[2] = {(unsigned)(uintptr_t)&Vs[0][0],
                      (unsigned)(uintptr_t)&Vs[1][0]};

  const int nch = 2 * qb + 2;                      // 64-key chunks for block

  // prologue: stage chunk 0 into buffer 0
  if (wave == 0) {
    tdm_load_2d(ldsK[0], Kh, 64, 64, D_, D_, S_, true);
    tdm_load_2d(ldsV[0], Vh, 64, 64, S_, S_, D_, true);
    __builtin_amdgcn_s_wait_tensorcnt(0);
  }
  __syncthreads();

  for (int c = 0; c < nch; ++c) {
    const int t0 = c * 64;
    const int buf = c & 1;

    // kick off chunk c+1 DMA into the other buffer (overlaps compute below)
    if (wave == 0 && c + 1 < nch) {
      tdm_load_2d(ldsK[buf ^ 1], Kh + (size_t)(t0 + 64) * D_, 64, 64, D_, D_, S_, true);
      tdm_load_2d(ldsV[buf ^ 1], Vh + (t0 + 64), 64, 64, S_, S_, D_, true);
    }

    if (t0 <= qr0 + 15) {                          // causal: wave has work
      // ---- scores: 16 rows x 64 keys ----
      v8f s[4];
#pragma unroll
      for (int jt = 0; jt < 4; ++jt) {
        v8f acc = {};
        const u16* kb = &Ks[buf][(size_t)(16 * jt) * LDP];
        acc = wmma_bf16(aq0, load_frag_b(kb, LDP, lane), acc);
        acc = wmma_bf16(aq1, load_frag_b(kb + 32, LDP, lane), acc);
        s[jt] = acc;
      }

      // ---- online softmax in exp2 domain (alpha folded in-loop) ----
#pragma unroll
      for (int v = 0; v < 8; ++v) {
        const int row = qr0 + v + 8 * half;
        float x0 = s[0][v] * SCL; if (t0 + n0 > row)      x0 = NEG;
        float x1 = s[1][v] * SCL; if (t0 + 16 + n0 > row) x1 = NEG;
        float x2 = s[2][v] * SCL; if (t0 + 32 + n0 > row) x2 = NEG;
        float x3 = s[3][v] * SCL; if (t0 + 48 + n0 > row) x3 = NEG;
        float mx = red16_max(fmaxf(fmaxf(x0, x1), fmaxf(x2, x3)));
        const float mn = fmaxf(mrow[v], mx);
        const float p0 = exp2f(x0 - mn);
        const float p1 = exp2f(x1 - mn);
        const float p2 = exp2f(x2 - mn);
        const float p3 = exp2f(x3 - mn);
        const float sum = red16_sum((p0 + p1) + (p2 + p3));
        const float alpha = exp2f(mrow[v] - mn);
        lrow[v] = lrow[v] * alpha + sum;
        mrow[v] = mn;
        o[0][v] *= alpha;
        o[1][v] *= alpha;
        o[2][v] *= alpha;
        o[3][v] *= alpha;
        u16* pr = &Pt[wave][(size_t)(v + 8 * half) * LDP];
        pr[n0]      = f2bf(p0);
        pr[n0 + 16] = f2bf(p1);
        pr[n0 + 32] = f2bf(p2);
        pr[n0 + 48] = f2bf(p3);
      }

      // wave-private LDS round-trip: wait for DS stores before fragment reads
      asm volatile("s_wait_dscnt 0x0" ::: "memory");

      // ---- PV: P(16x64) * V-cols (64 x 64) ----
      v16bf pa0 = load_frag_a(&Pt[wave][0], LDP, lane);
      v16bf pa1 = load_frag_a(&Pt[wave][32], LDP, lane);
#pragma unroll
      for (int j = 0; j < 4; ++j) {
        const u16* vb = &Vs[buf][(size_t)(16 * j) * LDP];
        o[j] = wmma_bf16(pa0, load_frag_b(vb, LDP, lane), o[j]);
        o[j] = wmma_bf16(pa1, load_frag_b(vb + 32, LDP, lane), o[j]);
      }
    }

    // chunk c+1 DMA must be complete before anyone reads it next iteration
    if (wave == 0) __builtin_amdgcn_s_wait_tensorcnt(0);
    __syncthreads();
  }

  // ---- epilogue: normalize, store bf16 attention output [S][D] ----
#pragma unroll
  for (int v = 0; v < 8; ++v) {
    const float inv = 1.0f / lrow[v];
    const int row = qr0 + v + 8 * half;
#pragma unroll
    for (int j = 0; j < 4; ++j) {
      const int col = h * HD_ + j * 16 + n0;
      Ab[(size_t)row * D_ + col] = f2bf(o[j][v] * inv);
    }
  }
}

// ---- launcher ------------------------------------------------------------

extern "C" void kernel_launch(void* const* d_in, const int* in_sizes, int n_in,
                              void* d_out, int out_size, void* d_ws, size_t ws_size,
                              hipStream_t stream) {
  const float* x  = (const float*)d_in[0];
  const float* Wq = (const float*)d_in[1];
  const float* Wk = (const float*)d_in[2];
  const float* Wv = (const float*)d_in[3];
  const float* Wo = (const float*)d_in[4];

  char* ws = (char*)d_ws;
  u16* xb  = (u16*)(ws);                         // 8 MiB
  u16* wqb = (u16*)(ws + (8u << 20));            // 2 MiB each
  u16* wkb = (u16*)(ws + (10u << 20));
  u16* wvb = (u16*)(ws + (12u << 20));
  u16* wob = (u16*)(ws + (14u << 20));
  u16* Qb  = (u16*)(ws + (16u << 20));           // 8 MiB each
  u16* Kb  = (u16*)(ws + (24u << 20));
  u16* Vtb = (u16*)(ws + (32u << 20));           // transposed V [e][s]
  u16* Abf = (u16*)(ws + (40u << 20));           // attention out, bf16

  cvt_f32_bf16<<<2048, 256, 0, stream>>>(x,  xb,  S_ * D_);
  cvt_f32_bf16<<<512,  256, 0, stream>>>(Wq, wqb, D_ * D_);
  cvt_f32_bf16<<<512,  256, 0, stream>>>(Wk, wkb, D_ * D_);
  cvt_f32_bf16<<<512,  256, 0, stream>>>(Wv, wvb, D_ * D_);
  cvt_f32_bf16<<<512,  256, 0, stream>>>(Wo, wob, D_ * D_);

  gemm16<0><<<512, 256, 0, stream>>>(xb, wqb, Qb);
  gemm16<0><<<512, 256, 0, stream>>>(xb, wkb, Kb);
  gemm16<1><<<512, 256, 0, stream>>>(xb, wvb, Vtb);

  flash_attn<<<512, 256, 0, stream>>>(Qb, Kb, Vtb, Abf);

  gemm16<2><<<512, 256, 0, stream>>>(Abf, wob, d_out);
}
